// ModifiedIBRNet_6468220748373
// MI455X (gfx1250) — matrix operand order repaired and checked by reading, not compiled, not run on hardware
//
#include <hip/hip_runtime.h>

typedef __attribute__((ext_vector_type(16))) _Float16 half16;
typedef __attribute__((ext_vector_type(8)))  _Float16 half8;
typedef __attribute__((ext_vector_type(8)))  float    float8;

// ---------------- tile table (16x16 N  x 32 K f16 B-tiles, 1KB each) ----------
constexpr int T_BW1 = 0;   // base_w1 57x64  : 2 kt x 4 nt = 8
constexpr int T_BW2 = 8;   // base_w2 64x32  : 2 x 2 = 4
constexpr int T_VF1 = 12;  // visf_w1 32x32  : 1 x 2 = 2
constexpr int T_VF2 = 14;  // visf_w2 32x33  : 1 x 2 = 2 (col 32 handled VALU)
constexpr int T_V21 = 16;  // vis2_w1 32x32  : 1 x 2 = 2
constexpr int T_GW1 = 18;  // geo_w1 65x64   : 2 x 4 = 8 (row 64 rank-1 VALU)
constexpr int T_GW2 = 26;  // geo_w2 64x16   : 2 x 1 = 2
constexpr int T_RW1 = 28;  // rgb_w1 33x16   : 2 x 1 = 2
constexpr int T_RW2 = 30;  // rgb_w2 16x8    : 1 x 1 = 1
constexpr int NTILES_W = 31;

// ---------------- const-buffer float offsets ---------------------------------
constexpr int CB_B1=0, CB_B2=64, CB_VB1=96, CB_VB2=128, CB_VF2C=161, CB_V2B1=193,
              CB_V2W2=225, CB_V2B2=257, CB_GB1=258, CB_GB2=322, CB_GR64=338,
              CB_WQ=402, CB_WK=658, CB_WV=914, CB_WO=1170, CB_LNG=1426,
              CB_LNB=1442, CB_OG1=1458, CB_OGB1=1714, CB_OGW2=1730, CB_OGB2=1746,
              CB_RB1=1747, CB_RB2=1763, CB_RW3=1771, CB_RB3=1779, CB_POS=1780,
              CB_TOTAL=1844;

// ---------------- LDS layout (bytes) ------------------------------------------
constexpr size_t O_WT    = 0;
constexpr size_t O_GFIN  = O_WT    + (size_t)NTILES_W*512*2; // 64 pairs x 64 f16
constexpr size_t O_IN    = O_GFIN  + 64*64*2;                // 4 waves x 16x64 f16
constexpr size_t O_CB    = O_IN    + 4*16*64*2;
constexpr size_t O_GFEAT = O_CB    + ((CB_TOTAL+3)&~3)*4;    // 64 x 38 f32
constexpr size_t O_WROW  = O_GFEAT + 64*38*4;                // 512 f32
constexpr size_t O_MASKB = O_WROW  + 512*4;
constexpr size_t O_VISB  = O_MASKB + 512*4;
constexpr size_t O_LOGB  = O_VISB  + 512*4;
constexpr size_t O_NVAL  = O_LOGB  + 512*4;                  // 64
constexpr size_t O_MW    = O_NVAL  + 64*4;                   // 64
constexpr size_t O_SIG   = O_MW    + 64*4;                   // 64
constexpr size_t O_ROWT  = O_SIG   + 64*4;                   // 4x16
constexpr size_t O_XBUF  = O_ROWT  + 64*4;                   // 512 x 32 f32
constexpr size_t O_GFBUF = O_XBUF  + (size_t)512*32*4;       // 64 x 16 f32
constexpr size_t O_KV    = O_GFBUF + 64*16*4;                // 64 x 32 f32
constexpr size_t SMEM_BYTES = O_KV + 64*32*4;                // ~149 KB

struct Params {
  const float *rgb, *mask;
  const float *bw1,*bb1,*bw2,*bb2;
  const float *vf1,*vfb1,*vf2,*vfb2;
  const float *v21,*v2b1,*v2w2,*v2b2;
  const float *gw1,*gb1,*gw2,*gb2;
  const float *wq,*wk,*wv,*wo,*lng,*lnb;
  const float *og1,*ogb1,*ogw2,*ogb2;
  const float *rw1,*rb1,*rw2,*rb2,*rw3,*rb3;
  float* out;
};

__device__ inline float eluf(float x){ return x > 0.f ? x : (__expf(x) - 1.f); }
__device__ inline float sigm(float x){ return 1.f / (1.f + __expf(-x)); }

__device__ inline float8 wmma32(half16 a, half16 b, float8 c){
  return __builtin_amdgcn_wmma_f32_16x16x32_f16(false, a, false, b, (short)0, c,
                                                false, false);
}

__device__ inline half16 mk16(half8 lo, half8 hi){
  half16 r;
#pragma unroll
  for (int i = 0; i < 8; ++i){ r[i] = lo[i]; r[i+8] = hi[i]; }
  return r;
}

// A fragment: row = lane&15; lanes 0-15 hold K {0-7,16-23}, lanes 16-31 {8-15,24-31}
__device__ inline half16 loadA(const _Float16* base, int lane, int ck){
  const int row = lane & 15;
  const int kb  = (lane & 16) ? 8 : 0;
  const _Float16* p = base + row*64 + ck*32 + kb;
  return mk16(*(const half8*)p, *(const half8*)(p + 16));
}
// B fragment: col = lane&15; lanes 0-15 hold K {0-7,16-23}, lanes 16-31 {8-15,24-31}
__device__ inline half16 loadB(const _Float16* wt, int tile, int lane){
  const _Float16* p = wt + (size_t)tile*512 + lane*16;
  return mk16(*(const half8*)p, *(const half8*)(p + 8));
}

__device__ inline void pack_weight(const float* W, int Krows, int Ncols,
                                   int ktiles, int ntiles, _Float16* dst, int tid){
  const int total = ktiles*ntiles*32;
  for (int idx = tid; idx < total; idx += 128){
    const int lane = idx & 31, tile = idx >> 5;
    const int kt = tile / ntiles, nt = tile % ntiles;
    _Float16* d = dst + (size_t)tile*512 + lane*16;
    const int n  = nt*16 + (lane & 15);
    const int kb = kt*32 + ((lane & 16) ? 8 : 0);
#pragma unroll
    for (int i = 0; i < 16; ++i){
      const int k = kb + (i < 8 ? i : i + 8);
      float v = (k < Krows && n < Ncols) ? W[k*Ncols + n] : 0.f;
      d[i] = (_Float16)v;
    }
  }
}

__device__ inline void cpf(float* dst, const float* src, int n, int tid){
  for (int i = tid; i < n; i += 128) dst[i] = src[i];
}

__global__ __launch_bounds__(128)
void ibrnet_fused_kernel(Params P){
  extern __shared__ char smem[];
  _Float16* WT    = (_Float16*)(smem + O_WT);
  _Float16* GFIN  = (_Float16*)(smem + O_GFIN);
  _Float16* INB   = (_Float16*)(smem + O_IN);
  float* CB    = (float*)(smem + O_CB);
  float* GFEAT = (float*)(smem + O_GFEAT);
  float* WROW  = (float*)(smem + O_WROW);
  float* MASKB = (float*)(smem + O_MASKB);
  float* VISB  = (float*)(smem + O_VISB);
  float* LOGB  = (float*)(smem + O_LOGB);
  float* NVAL  = (float*)(smem + O_NVAL);
  float* MW    = (float*)(smem + O_MW);
  float* SIG   = (float*)(smem + O_SIG);
  float* ROWT  = (float*)(smem + O_ROWT);
  float* XBUF  = (float*)(smem + O_XBUF);
  float* GFBUF = (float*)(smem + O_GFBUF);
  float* KV    = (float*)(smem + O_KV);

  const int tid  = threadIdx.x;
  const int wave = tid >> 5;
  const int lane = tid & 31;
  const int cn   = lane & 15;
  const int rh   = (lane & 16) ? 8 : 0;
  _Float16* in = INB + wave*1024;

  // ---------------- phase 0: pack weights/consts to LDS -----------------------
  pack_weight(P.bw1, 57, 64, 2, 4, WT + (size_t)T_BW1*512, tid);
  pack_weight(P.bw2, 64, 32, 2, 2, WT + (size_t)T_BW2*512, tid);
  pack_weight(P.vf1, 32, 32, 1, 2, WT + (size_t)T_VF1*512, tid);
  pack_weight(P.vf2, 32, 33, 1, 2, WT + (size_t)T_VF2*512, tid);
  pack_weight(P.v21, 32, 32, 1, 2, WT + (size_t)T_V21*512, tid);
  pack_weight(P.gw1, 65, 64, 2, 4, WT + (size_t)T_GW1*512, tid);
  pack_weight(P.gw2, 64, 16, 2, 1, WT + (size_t)T_GW2*512, tid);
  pack_weight(P.rw1, 33, 16, 2, 1, WT + (size_t)T_RW1*512, tid);
  pack_weight(P.rw2, 16,  8, 1, 1, WT + (size_t)T_RW2*512, tid);

  cpf(CB+CB_B1,  P.bb1, 64, tid);  cpf(CB+CB_B2,  P.bb2, 32, tid);
  cpf(CB+CB_VB1, P.vfb1,32, tid);  cpf(CB+CB_VB2, P.vfb2,33, tid);
  cpf(CB+CB_V2B1,P.v2b1,32, tid);
  cpf(CB+CB_GB1, P.gb1, 64, tid);  cpf(CB+CB_GB2, P.gb2, 16, tid);
  cpf(CB+CB_WQ,  P.wq, 256, tid);  cpf(CB+CB_WK,  P.wk, 256, tid);
  cpf(CB+CB_WV,  P.wv, 256, tid);  cpf(CB+CB_WO,  P.wo, 256, tid);
  cpf(CB+CB_LNG, P.lng, 16, tid);  cpf(CB+CB_LNB, P.lnb, 16, tid);
  cpf(CB+CB_OG1, P.og1,256, tid);  cpf(CB+CB_OGB1,P.ogb1,16, tid);
  cpf(CB+CB_OGW2,P.ogw2,16, tid);
  cpf(CB+CB_RB1, P.rb1, 16, tid);  cpf(CB+CB_RB2, P.rb2,  8, tid);
  cpf(CB+CB_RW3, P.rw3,  8, tid);
  for (int i = tid; i < 32; i += 128) CB[CB_VF2C+i] = P.vf2[i*33 + 32];
  for (int i = tid; i < 32; i += 128) CB[CB_V2W2+i] = P.v2w2[i];
  for (int i = tid; i < 64; i += 128) CB[CB_GR64+i] = P.gw1[64*64 + i];
  if (tid == 0){
    CB[CB_V2B2] = P.v2b2[0]; CB[CB_OGB2] = P.ogb2[0]; CB[CB_RB3] = P.rb3[0];
  }
  if (tid < 64){ // posenc(16,4)
    int pos = tid >> 4, j = tid & 15;
    float e = 2.f * (float)(j >> 1) / 16.f;
    float ang = (float)pos / __expf(e * 9.210340371976184f);
    CB[CB_POS + tid] = (j & 1) ? cosf(ang) : sinf(ang);
  }
  __syncthreads();

  // ---------------- phase 1: per-(n,s) weights + weighted mean/var ------------
  if (tid < 64){
    const int p = tid;
    const long gp = (long)blockIdx.x*64 + p;
    float ms = 0.f, mv[8];
#pragma unroll
    for (int v = 0; v < 8; ++v){ mv[v] = P.mask[gp*8 + v]; MASKB[p*8+v] = mv[v]; ms += mv[v]; }
    NVAL[p] = ms;
    const float inv = 1.f / (ms + 1e-8f);
#pragma unroll
    for (int v = 0; v < 8; ++v) WROW[p*8+v] = mv[v]*inv;
    for (int f = 0; f < 19; ++f){
      float mu = 0.f;
#pragma unroll
      for (int v = 0; v < 8; ++v) mu += WROW[p*8+v]*P.rgb[(gp*8+v)*19 + f];
      float va = 0.f;
#pragma unroll
      for (int v = 0; v < 8; ++v){ float d = P.rgb[(gp*8+v)*19 + f] - mu; va += WROW[p*8+v]*d*d; }
      GFEAT[p*38 + f] = mu; GFEAT[p*38 + 19 + f] = va;
    }
  }
  __syncthreads();

  // ---------------- phase 2: stage-A MLP chain (WMMA) -------------------------
  // Hoist all 18 loop-invariant B fragments (tiles 0..17) into registers: the
  // compiler cannot do this itself because the loop stores to LDS and it can't
  // prove non-aliasing with the weight region. Saves ~36 ds_load_b128 per tile.
  {
    half16 Bw[18];
#pragma unroll
    for (int i = 0; i < 18; ++i) Bw[i] = loadB(WT, i, lane);

    for (int t = wave; t < 32; t += 4){
      const int row0 = t*16;
      { // build 57-wide input (pad to 64), 2 lanes per row
        const int m = lane & 15, ck = lane >> 4;
        const int lr = row0 + m, p = lr >> 3;
        const long g = (long)blockIdx.x*512 + lr;
#pragma unroll
        for (int j = 0; j < 32; ++j){
          const int col = ck*32 + j; float v;
          if      (col < 38) v = GFEAT[p*38 + col];
          else if (col < 57) v = P.rgb[g*19 + (col - 38)];
          else               v = 0.f;
          in[m*64 + col] = (_Float16)v;
        }
      }
      half16 A0 = loadA(in, lane, 0), A1 = loadA(in, lane, 1);
#pragma unroll
      for (int nt = 0; nt < 4; ++nt){                     // base L1: 57->64
        float8 c = {};
        c = wmma32(A0, Bw[T_BW1 + nt],     c);
        c = wmma32(A1, Bw[T_BW1 + 4 + nt], c);
        const float b = CB[CB_B1 + nt*16 + cn];
#pragma unroll
        for (int r = 0; r < 8; ++r) in[(r+rh)*64 + nt*16 + cn] = (_Float16)eluf(c[r] + b);
      }
      A0 = loadA(in, lane, 0); A1 = loadA(in, lane, 1);
      float8 x0[2];
#pragma unroll
      for (int nt = 0; nt < 2; ++nt){                     // base L2: 64->32
        float8 c = {};
        c = wmma32(A0, Bw[T_BW2 + nt],     c);
        c = wmma32(A1, Bw[T_BW2 + 2 + nt], c);
        const float b = CB[CB_B2 + nt*16 + cn];
#pragma unroll
        for (int r = 0; r < 8; ++r){
          const float x = eluf(c[r] + b);
          x0[nt][r] = x;
          in[(r+rh)*64 + nt*16 + cn] = (_Float16)(x * WROW[row0 + r + rh]); // x*w
        }
      }
      A0 = loadA(in, lane, 0);
#pragma unroll
      for (int nt = 0; nt < 2; ++nt){                     // visf L1: 32->32
        float8 c = {};
        c = wmma32(A0, Bw[T_VF1 + nt], c);
        const float b = CB[CB_VB1 + nt*16 + cn];
#pragma unroll
        for (int r = 0; r < 8; ++r) in[(r+rh)*64 + nt*16 + cn] = (_Float16)eluf(c[r] + b);
      }
      float vlog = 0.f;
      if (lane < 16){                                     // visf L2 col 32 (vis logit)
        vlog = CB[CB_VB2 + 32];
        for (int kk = 0; kk < 32; ++kk) vlog += (float)in[lane*64 + kk]*CB[CB_VF2C + kk];
      }
      A0 = loadA(in, lane, 0);
#pragma unroll
      for (int nt = 0; nt < 2; ++nt){                     // visf L2 cols 0..31 (x_res)
        float8 c = {};
        c = wmma32(A0, Bw[T_VF2 + nt], c);
        const float b = CB[CB_VB2 + nt*16 + cn];
#pragma unroll
        for (int r = 0; r < 8; ++r) x0[nt][r] += eluf(c[r] + b);   // residual
      }
      if (lane < 16) ROWT[wave*16 + lane] = sigm(vlog) * MASKB[row0 + lane];
#pragma unroll
      for (int nt = 0; nt < 2; ++nt)
#pragma unroll
        for (int r = 0; r < 8; ++r){
          const int lr = row0 + r + rh;
          const float x = x0[nt][r];
          XBUF[lr*32 + nt*16 + cn] = x;
          in[(r+rh)*64 + nt*16 + cn] = (_Float16)(x * ROWT[wave*16 + r + rh]); // x*vis
        }
      A0 = loadA(in, lane, 0);
#pragma unroll
      for (int nt = 0; nt < 2; ++nt){                     // vis2 L1: 32->32
        float8 c = {};
        c = wmma32(A0, Bw[T_V21 + nt], c);
        const float b = CB[CB_V2B1 + nt*16 + cn];
#pragma unroll
        for (int r = 0; r < 8; ++r) in[(r+rh)*64 + nt*16 + cn] = (_Float16)eluf(c[r] + b);
      }
      if (lane < 16){                                     // vis2 L2: 32->1
        float a = CB[CB_V2B2];
        for (int kk = 0; kk < 32; ++kk) a += (float)in[lane*64 + kk]*CB[CB_V2W2 + kk];
        VISB[row0 + lane] = sigm(a) * MASKB[row0 + lane];
      }
    }
  }
  __syncthreads();

  // ---------------- phase 3a: per-pair weighted stats over V ------------------
  if (tid < 64){
    const int p = tid;
    float vs = 0.f;
#pragma unroll
    for (int v = 0; v < 8; ++v) vs += VISB[p*8+v];
    const float inv = 1.f / (vs + 1e-8f);
    float w2[8];
#pragma unroll
    for (int v = 0; v < 8; ++v) w2[v] = VISB[p*8+v]*inv;
    MW[p] = vs*inv*0.125f;
    for (int c = 0; c < 32; ++c){
      float mu = 0.f;
#pragma unroll
      for (int v = 0; v < 8; ++v) mu += w2[v]*XBUF[(p*8+v)*32 + c];
      float va = 0.f;
#pragma unroll
      for (int v = 0; v < 8; ++v){ float d = XBUF[(p*8+v)*32 + c] - mu; va += w2[v]*d*d; }
      GFIN[p*64 + c] = (_Float16)mu;
      GFIN[p*64 + 32 + c] = (_Float16)va;
    }
  }
  __syncthreads();

  // ---------------- phase 3b: geo MLP (WMMA) + posenc -------------------------
  {
    const _Float16* gin = GFIN + wave*16*64;
    half16 A0 = loadA(gin, lane, 0), A1 = loadA(gin, lane, 1);
#pragma unroll
    for (int nt = 0; nt < 4; ++nt){                       // geo L1: 65->64
      float8 c = {};
      c = wmma32(A0, loadB(WT, T_GW1 + nt,     lane), c);
      c = wmma32(A1, loadB(WT, T_GW1 + 4 + nt, lane), c);
      const float wr = CB[CB_GR64 + nt*16 + cn];
      const float b  = CB[CB_GB1  + nt*16 + cn];
#pragma unroll
      for (int r = 0; r < 8; ++r){
        const int p = wave*16 + r + rh;
        in[(r+rh)*64 + nt*16 + cn] = (_Float16)eluf(c[r] + b + MW[p]*wr);
      }
    }
    A0 = loadA(in, lane, 0); A1 = loadA(in, lane, 1);
    float8 c = {};
    c = wmma32(A0, loadB(WT, T_GW2 + 0, lane), c);        // geo L2: 64->16
    c = wmma32(A1, loadB(WT, T_GW2 + 1, lane), c);
    const float b = CB[CB_GB2 + cn];
#pragma unroll
    for (int r = 0; r < 8; ++r){
      const int p = wave*16 + r + rh;
      GFBUF[p*16 + cn] = eluf(c[r] + b) + CB[CB_POS + (p & 3)*16 + cn];
    }
  }
  __syncthreads();

  // ---------------- phase 4: S=4, H=4 attention + LN + sigma head -------------
  if (tid < 64){
    const int p = tid;
    float gf[16];
#pragma unroll
    for (int i = 0; i < 16; ++i) gf[i] = GFBUF[p*16 + i];
#pragma unroll
    for (int i = 0; i < 16; ++i){
      float a = 0.f, b = 0.f;
#pragma unroll
      for (int j = 0; j < 16; ++j){ a += gf[j]*CB[CB_WK + j*16 + i]; b += gf[j]*CB[CB_WV + j*16 + i]; }
      KV[p*32 + i] = a; KV[p*32 + 16 + i] = b;
    }
  }
  __syncthreads();
  if (tid < 64){
    const int p = tid, nb = p & ~3;
    float gf[16], q[16], o[16];
#pragma unroll
    for (int i = 0; i < 16; ++i) gf[i] = GFBUF[p*16 + i];
#pragma unroll
    for (int i = 0; i < 16; ++i){
      float a = 0.f;
#pragma unroll
      for (int j = 0; j < 16; ++j) a += gf[j]*CB[CB_WQ + j*16 + i];
      q[i] = a;
    }
    const bool valid = NVAL[p] > 1.f;
#pragma unroll
    for (int h = 0; h < 4; ++h){
      float sc[4], mx = -1e30f;
#pragma unroll
      for (int s2 = 0; s2 < 4; ++s2){
        float d = 0.f;
#pragma unroll
        for (int dd = 0; dd < 4; ++dd) d += q[h*4+dd]*KV[(nb+s2)*32 + h*4 + dd];
        sc[s2] = d*0.5f; mx = fmaxf(mx, sc[s2]);
      }
      float pr[4];
      if (valid){
        float se = 0.f;
#pragma unroll
        for (int s2 = 0; s2 < 4; ++s2){ pr[s2] = __expf(sc[s2]-mx); se += pr[s2]; }
        const float iv = 1.f/se;
#pragma unroll
        for (int s2 = 0; s2 < 4; ++s2) pr[s2] *= iv;
      } else {
#pragma unroll
        for (int s2 = 0; s2 < 4; ++s2) pr[s2] = 0.25f;
      }
#pragma unroll
      for (int dd = 0; dd < 4; ++dd){
        float a = 0.f;
#pragma unroll
        for (int s2 = 0; s2 < 4; ++s2) a += pr[s2]*KV[(nb+s2)*32 + 16 + h*4 + dd];
        o[h*4+dd] = a;
      }
    }
    float oo[16], mu = 0.f;
#pragma unroll
    for (int i = 0; i < 16; ++i){
      float a = gf[i];
#pragma unroll
      for (int j = 0; j < 16; ++j) a += o[j]*CB[CB_WO + j*16 + i];
      oo[i] = a; mu += a;
    }
    mu *= (1.f/16.f);
    float va = 0.f;
#pragma unroll
    for (int i = 0; i < 16; ++i){ float d = oo[i]-mu; va += d*d; }
    va *= (1.f/16.f);
    const float is = rsqrtf(va + 1e-6f);
    float gn[16];
#pragma unroll
    for (int i = 0; i < 16; ++i) gn[i] = (oo[i]-mu)*is*CB[CB_LNG+i] + CB[CB_LNB+i];
    float sg = CB[CB_OGB2];
#pragma unroll
    for (int i = 0; i < 16; ++i){
      float a = CB[CB_OGB1 + i];
#pragma unroll
      for (int j = 0; j < 16; ++j) a += gn[j]*CB[CB_OG1 + j*16 + i];
      sg += eluf(a)*CB[CB_OGW2 + i];
    }
    SIG[p] = fmaxf(sg, 0.f);
  }
  __syncthreads();

  // ---------------- phase 5: rgb-blend MLP (WMMA) -----------------------------
  {
    half16 Br0 = loadB(WT, T_RW1 + 0, lane);
    half16 Br1 = loadB(WT, T_RW1 + 1, lane);
    half16 Br2 = loadB(WT, T_RW2,     lane);
    for (int t = wave; t < 32; t += 4){
      const int row0 = t*16;
      {
        const int m = lane & 15, ck = lane >> 4;
        const int lr = row0 + m;
#pragma unroll
        for (int j = 0; j < 32; ++j){
          const int col = ck*32 + j;
          float v = (col < 32) ? XBUF[lr*32 + col] : (col == 32 ? VISB[lr] : 0.f);
          in[m*64 + col] = (_Float16)v;
        }
      }
      half16 A0 = loadA(in, lane, 0), A1 = loadA(in, lane, 1);
      float8 c = {};
      c = wmma32(A0, Br0, c);                             // rgb L1: 33->16
      c = wmma32(A1, Br1, c);
      const float b = CB[CB_RB1 + cn];
#pragma unroll
      for (int r = 0; r < 8; ++r) in[(r+rh)*64 + cn] = (_Float16)eluf(c[r] + b);
      A0 = loadA(in, lane, 0);
      float8 c2 = {};
      c2 = wmma32(A0, Br2, c2);                           // rgb L2: 16->8
      if (cn < 8){
        const float b2 = CB[CB_RB2 + cn];
#pragma unroll
        for (int r = 0; r < 8; ++r) in[(r+rh)*64 + cn] = (_Float16)eluf(c2[r] + b2);
      }
      if (lane < 16){                                     // rgb L3: 8->1
        float a = CB[CB_RB3];
#pragma unroll
        for (int j = 0; j < 8; ++j) a += (float)in[lane*64 + j]*CB[CB_RW3 + j];
        LOGB[row0 + lane] = a;
      }
    }
  }
  __syncthreads();

  // ---------------- phase 6: masked softmax blend over V + output -------------
  if (tid < 64){
    const int p = tid;
    const long gp = (long)blockIdx.x*64 + p;
    float lg[8], mx = -1e30f;
#pragma unroll
    for (int v = 0; v < 8; ++v){
      const float m = MASKB[p*8+v];
      lg[v] = (m == 0.f) ? -1e9f : LOGB[p*8+v];
      mx = fmaxf(mx, lg[v]);
    }
    float se = 0.f;
#pragma unroll
    for (int v = 0; v < 8; ++v){ lg[v] = __expf(lg[v]-mx); se += lg[v]; }
    const float iv = 1.f/se;
    float r0 = 0.f, r1 = 0.f, r2 = 0.f;
#pragma unroll
    for (int v = 0; v < 8; ++v){
      const float bw = lg[v]*iv;
      const float* rf = P.rgb + (gp*8 + v)*19;
      r0 += bw*rf[0]; r1 += bw*rf[1]; r2 += bw*rf[2];
    }
    P.out[gp*4 + 0] = r0; P.out[gp*4 + 1] = r1; P.out[gp*4 + 2] = r2;
    P.out[gp*4 + 3] = SIG[p];
  }
}

extern "C" void kernel_launch(void* const* d_in, const int* in_sizes, int n_in,
                              void* d_out, int out_size, void* d_ws, size_t ws_size,
                              hipStream_t stream) {
  (void)in_sizes; (void)n_in; (void)out_size; (void)d_ws; (void)ws_size;
  Params P;
  P.rgb  = (const float*)d_in[0];  P.mask = (const float*)d_in[1];
  P.bw1  = (const float*)d_in[2];  P.bb1  = (const float*)d_in[3];
  P.bw2  = (const float*)d_in[4];  P.bb2  = (const float*)d_in[5];
  P.vf1  = (const float*)d_in[6];  P.vfb1 = (const float*)d_in[7];
  P.vf2  = (const float*)d_in[8];  P.vfb2 = (const float*)d_in[9];
  P.v21  = (const float*)d_in[10]; P.v2b1 = (const float*)d_in[11];
  P.v2w2 = (const float*)d_in[12]; P.v2b2 = (const float*)d_in[13];
  P.gw1  = (const float*)d_in[14]; P.gb1  = (const float*)d_in[15];
  P.gw2  = (const float*)d_in[16]; P.gb2  = (const float*)d_in[17];
  P.wq   = (const float*)d_in[18]; P.wk   = (const float*)d_in[19];
  P.wv   = (const float*)d_in[20]; P.wo   = (const float*)d_in[21];
  P.lng  = (const float*)d_in[22]; P.lnb  = (const float*)d_in[23];
  P.og1  = (const float*)d_in[24]; P.ogb1 = (const float*)d_in[25];
  P.ogw2 = (const float*)d_in[26]; P.ogb2 = (const float*)d_in[27];
  P.rw1  = (const float*)d_in[28]; P.rb1  = (const float*)d_in[29];
  P.rw2  = (const float*)d_in[30]; P.rb2  = (const float*)d_in[31];
  P.rw3  = (const float*)d_in[32]; P.rb3  = (const float*)d_in[33];
  P.out  = (float*)d_out;

  (void)hipFuncSetAttribute((const void*)ibrnet_fused_kernel,
                            hipFuncAttributeMaxDynamicSharedMemorySize,
                            (int)SMEM_BYTES);
  const int blocks = 65536 / 16;  // 16 rays per block, N=65536
  ibrnet_fused_kernel<<<dim3(blocks), dim3(128), SMEM_BYTES, stream>>>(P);
}
